// UIQueryMixer_86998857548016
// MI455X (gfx1250) — compile-verified, matrix operand order checked by reading, not run on hardware
//
#include <hip/hip_runtime.h>

// ---------------------------------------------------------------------------
// UIQueryMixer for MI455X (gfx1250, wave32, WMMA, TDM).
// bf16 matmul inputs, f32 accumulation (V_WMMA_F32_16X16X32_BF16),
// f32 residual/normalization path; p-tile staged into LDS via the
// Tensor Data Mover (tensor_load_to_lds + s_wait_tensorcnt).
// ---------------------------------------------------------------------------

typedef __attribute__((ext_vector_type(16))) __bf16 v16bf;
typedef __attribute__((ext_vector_type(8)))  float  v8f;
typedef v16bf v16bf_u __attribute__((aligned(16)));
typedef unsigned int uint32;
typedef __attribute__((ext_vector_type(4))) uint32 u32x4;
typedef __attribute__((ext_vector_type(8))) uint32 u32x8;

#define BATCHN 8192
#define NH     16
#define HD     256
#define FH     512
#define EPSV   1e-6f

#define KT1 8    // HD/32  (k-tiles, GEMM1)
#define NT1 32   // FH/16  (n-tiles, GEMM1)
#define KT2 16   // FH/32  (k-tiles, GEMM2)
#define NT2 16   // HD/16  (n-tiles, GEMM2)

#define BM   64          // batch rows per workgroup tile
#define PSTR (HD + 8)    // f32 LDS stride (bank rotate; matches TDM pad 8 dw)
#define ASTR (HD + 8)    // bf16 LDS stride
#define GSTR (FH + 8)    // bf16 LDS stride

// ---------------------------------------------------------------------------
// K0: pack f32 weights into bf16, fragment-linear layout.
// Fragment = 32x16 (KxN) B-tile; lane L holds col=L%16, k-half=L/16, 16 contig
// k-values -> each lane's 16 bf16 (32B) stored contiguously, lane-major.
// dst flat index = (((n*KT + kt)*NT + nt)*32 + lane)*16 + e
// ---------------------------------------------------------------------------
__global__ void pack_weights(const float* __restrict__ w1,
                             const float* __restrict__ w3,
                             const float* __restrict__ w2,
                             __bf16* __restrict__ w1p,
                             __bf16* __restrict__ w3p,
                             __bf16* __restrict__ w2p) {
  int which = blockIdx.y;                       // 0: w1, 1: w3, 2: w2
  int id   = blockIdx.x * blockDim.x + threadIdx.x;  // 0..131071
  int lane = id & 31;
  int slot = id >> 5;                           // (n*KT + kt)*NT + nt
  if (which < 2) {
    const float* src = (which == 0) ? w1 : w3;  // (NH, HD, FH) row-major
    __bf16* dst = (which == 0) ? w1p : w3p;
    int nt = slot % NT1; int tmp = slot / NT1;
    int kt = tmp % KT1;  int n  = tmp / KT1;
    int col = nt * 16 + (lane & 15);
    int k0  = kt * 32 + (lane >> 4) * 16;
    const float* s = src + ((size_t)n * HD + k0) * FH + col;
    __bf16* d = dst + (size_t)id * 16;
#pragma unroll
    for (int e = 0; e < 16; ++e) d[e] = (__bf16)s[(size_t)e * FH];
  } else {
    // w2: (NH, FH, HD) row-major; K = FH, N = HD
    int nt = slot % NT2; int tmp = slot / NT2;
    int kt = tmp % KT2;  int n  = tmp / KT2;
    int col = nt * 16 + (lane & 15);
    int k0  = kt * 32 + (lane >> 4) * 16;
    const float* s = w2 + ((size_t)n * FH + k0) * HD + col;
    __bf16* d = w2p + (size_t)id * 16;
#pragma unroll
    for (int e = 0; e < 16; ++e) d[e] = (__bf16)s[(size_t)e * HD];
  }
}

// ---------------------------------------------------------------------------
// K1: rmsnorm1 + cross-head sub-dim mixing + mask + residual -> p (f32).
// One 256-thread block per batch row. thread t -> head hd=t/16, chunk s=t%16
// (16 elements). Mixing is a (head, chunk) transpose done through padded LDS.
// ---------------------------------------------------------------------------
__global__ void mix_kernel(const float* __restrict__ x,
                           const float* __restrict__ n1w,
                           const float* __restrict__ mask,
                           float* __restrict__ p) {
  __shared__ float xn[NH][HD + 4];   // padded: bank-conflict-free transpose
  __shared__ float red[256];
  int b  = blockIdx.x;
  int t  = threadIdx.x;
  int hd = t >> 4;
  int s  = t & 15;

  float xv[16];
  {
    const float4* xr = (const float4*)(x + ((size_t)b * NH + hd) * HD + s * 16);
#pragma unroll
    for (int i = 0; i < 4; ++i) {
      float4 v = xr[i];
      xv[4 * i + 0] = v.x; xv[4 * i + 1] = v.y;
      xv[4 * i + 2] = v.z; xv[4 * i + 3] = v.w;
    }
  }
  float ss = 0.f;
#pragma unroll
  for (int i = 0; i < 16; ++i) ss += xv[i] * xv[i];
  red[t] = ss;
  __syncthreads();
  float tot = 0.f;
#pragma unroll
  for (int j = 0; j < 16; ++j) tot += red[(hd << 4) + j];
  float r1 = rsqrtf(tot * (1.f / HD) + EPSV);
#pragma unroll
  for (int i = 0; i < 16; ++i)
    xn[hd][(s << 4) + i] = xv[i] * r1 * n1w[(s << 4) + i];
  __syncthreads();

  // mixed[hd, s*16+i] = xn[s][hd*16+i] * mask[hd, s*16+i];  p = mixed + x
  float pv[16];
#pragma unroll
  for (int i = 0; i < 16; ++i) {
    float m = xn[s][(hd << 4) + i] * mask[hd * HD + (s << 4) + i];
    pv[i] = m + xv[i];
  }
  float4* pr = (float4*)(p + ((size_t)b * NH + hd) * HD + s * 16);
#pragma unroll
  for (int i = 0; i < 4; ++i) {
    float4 v;
    v.x = pv[4 * i + 0]; v.y = pv[4 * i + 1];
    v.z = pv[4 * i + 2]; v.w = pv[4 * i + 3];
    pr[i] = v;
  }
}

// ---------------------------------------------------------------------------
// K2: per-(64-row, head) tile.
//   Phase 0: TDM DMA of the p-tile (64 rows x 256 f32, row stride NH*HD)
//            straight into LDS, HW-padded to PSTR=264 dwords per row.
//   Phase 1: in-LDS rmsnorm2 -> bf16 A-tile.
//   Phase 2: GEMM1 (W1 & W3 fused K-loop) -> silu-gate -> bf16 G in LDS.
//   Phase 3: GEMM2 (W2) + residual from LDS p-tile.
// 8 waves; ~165 KB static LDS (CDNA5: 320 KB/WGP).
// ---------------------------------------------------------------------------
__global__ __launch_bounds__(256)
void ffn_kernel(const float* __restrict__ p,
                const float* __restrict__ n2w,
                const __bf16* __restrict__ w1p,
                const __bf16* __restrict__ w3p,
                const __bf16* __restrict__ w2p,
                float* __restrict__ out) {
  __shared__ float  pT[BM * PSTR];      // 67.6 KB  (f32 residual tile, TDM dst)
  __shared__ __bf16 hA[BM * ASTR];      // 33.8 KB  (rmsnorm'd A-tile)
  __shared__ __bf16 G [BM * GSTR];      // 66.6 KB  (gated hidden)
  __shared__ float  rms[BM];
  __shared__ float  partials[256];

  int mb   = blockIdx.x;
  int head = blockIdx.y;
  int t    = threadIdx.x;
  int lane = t & 31;
  int w    = t >> 5;
  size_t b0 = (size_t)mb * BM;

  // ---- Phase 0: Tensor Data Mover load of the p-tile into LDS ----
  // 2D D#: tile_dim0=256 (f32), tile_dim1=64 rows, tensor_dim0_stride=NH*HD
  // (skips the other heads' rows); pad 8 dwords after every 256 dwords so the
  // LDS image has the bank-rotating PSTR=264 row stride.
  if (t < 32) {                       // one wave issues the DMA (EXEC ignored)
    unsigned long long ga =
        (unsigned long long)(size_t)(p + (b0 * NH + (size_t)head) * HD);
    uint32 lds0 = (uint32)(size_t)(void*)&pT[0];
    u32x4 g0;
    g0[0] = 1u;                                  // count=1 (valid user D#)
    g0[1] = lds0;                                // lds_addr (bytes)
    g0[2] = (uint32)(ga & 0xFFFFFFFFu);          // global_addr[31:0]
    g0[3] = (uint32)((ga >> 32) & 0x01FFFFFFu)   // global_addr[56:32]
          | (2u << 30);                          // type = 2 ("image")
    u32x8 g1;
    g1[0] = (2u << 16)                           // data_size = 4 bytes
          | (1u << 20)                           // pad_enable
          | (7u << 22)                           // pad_interval = 256 dwords
          | (7u << 25);                          // pad_amount  = 8 dwords
    g1[1] = (uint32)HD << 16;                    // tensor_dim0 = 256
    g1[2] = 0xFFFFu << 16;                       // tensor_dim1 = 65535
    g1[3] = (uint32)HD << 16;                    // tile_dim0 = 256
    g1[4] = (uint32)BM;                          // tile_dim1 = 64
    g1[5] = (uint32)(NH * HD);                   // tensor_dim0_stride = 4096
    g1[6] = 0u;
    g1[7] = 0u;
    asm volatile("tensor_load_to_lds %0, %1" :: "s"(g0), "s"(g1) : "memory");
    __builtin_amdgcn_s_wait_tensorcnt(0);
  }
  __syncthreads();

  // ---- Phase 1: per-row sum of squares from LDS, rms, bf16 A-tile ----
  int row = t >> 2;            // 0..63
  int c0  = (t & 3) * 64;      // 0/64/128/192
  {
    float part = 0.f;
#pragma unroll 16
    for (int i = 0; i < 64; ++i) {
      float v = pT[row * PSTR + c0 + i];
      part += v * v;
    }
    partials[t] = part;
  }
  __syncthreads();
  if ((t & 3) == 0) {
    float tot = partials[t] + partials[t + 1] + partials[t + 2] + partials[t + 3];
    rms[row] = rsqrtf(tot * (1.f / HD) + EPSV);
  }
  __syncthreads();
  {
    float r = rms[row];
#pragma unroll 8
    for (int i = 0; i < 64; ++i) {
      int c = c0 + i;
      hA[row * ASTR + c] = (__bf16)(pT[row * PSTR + c] * r * n2w[c]);
    }
  }
  __syncthreads();

  // ---- Phase 2: GEMM1: h @ W1, h @ W3 (fused K-loop), silu-gate -> G ----
  for (int hb = 0; hb < 2; ++hb) {
    int cbase = hb * 256 + w * 32;          // wave's H-column base
    v8f acc1[4][2] = {};
    v8f acc3[4][2] = {};
    int ar = lane & 15;
    for (int kt = 0; kt < KT1; ++kt) {
      int ak = kt * 32 + (lane >> 4) * 16;
      v16bf af[4];
#pragma unroll
      for (int m = 0; m < 4; ++m)
        af[m] = *(const v16bf_u*)&hA[(m * 16 + ar) * ASTR + ak];
#pragma unroll
      for (int n = 0; n < 2; ++n) {
        int ntAbs = (cbase >> 4) + n;
        size_t off = ((((size_t)head * KT1 + kt) * NT1 + ntAbs) * 32 + lane) * 16;
        if (kt + 1 < KT1) {
          size_t offn = ((((size_t)head * KT1 + kt + 1) * NT1 + ntAbs) * 32 + lane) * 16;
          __builtin_prefetch(&w1p[offn], 0, 1);
          __builtin_prefetch(&w3p[offn], 0, 1);
        }
        v16bf b1 = *(const v16bf_u*)&w1p[off];
        v16bf b3 = *(const v16bf_u*)&w3p[off];
#pragma unroll
        for (int m = 0; m < 4; ++m) {
          acc1[m][n] = __builtin_amdgcn_wmma_f32_16x16x32_bf16(
              false, af[m], false, b1, (short)0, acc1[m][n], false, false);
          acc3[m][n] = __builtin_amdgcn_wmma_f32_16x16x32_bf16(
              false, af[m], false, b3, (short)0, acc3[m][n], false, false);
        }
      }
    }
    // epilogue: G = silu(a) * b, bf16
#pragma unroll
    for (int m = 0; m < 4; ++m)
#pragma unroll
      for (int n = 0; n < 2; ++n)
#pragma unroll
        for (int rr = 0; rr < 8; ++rr) {
          float a = acc1[m][n][rr];
          float g = (a / (1.f + __expf(-a))) * acc3[m][n][rr];
          int rrow = m * 16 + (lane >> 4) * 8 + rr;
          int ccol = cbase + n * 16 + (lane & 15);
          G[rrow * GSTR + ccol] = (__bf16)g;
        }
  }
  __syncthreads();

  // ---- Phase 3: GEMM2: G @ W2 + residual ----
  {
    int cb2 = w * 32;                        // wave's output D-column base
    v8f acc[4][2] = {};
    int ar = lane & 15;
    for (int kt = 0; kt < KT2; ++kt) {
      int ak = kt * 32 + (lane >> 4) * 16;
      v16bf af[4];
#pragma unroll
      for (int m = 0; m < 4; ++m)
        af[m] = *(const v16bf_u*)&G[(m * 16 + ar) * GSTR + ak];
#pragma unroll
      for (int n = 0; n < 2; ++n) {
        int ntAbs = (cb2 >> 4) + n;
        size_t off = ((((size_t)head * KT2 + kt) * NT2 + ntAbs) * 32 + lane) * 16;
        if (kt + 1 < KT2) {
          size_t offn = ((((size_t)head * KT2 + kt + 1) * NT2 + ntAbs) * 32 + lane) * 16;
          __builtin_prefetch(&w2p[offn], 0, 1);
        }
        v16bf bf = *(const v16bf_u*)&w2p[off];
#pragma unroll
        for (int m = 0; m < 4; ++m)
          acc[m][n] = __builtin_amdgcn_wmma_f32_16x16x32_bf16(
              false, af[m], false, bf, (short)0, acc[m][n], false, false);
      }
    }
#pragma unroll
    for (int m = 0; m < 4; ++m)
#pragma unroll
      for (int n = 0; n < 2; ++n)
#pragma unroll
        for (int rr = 0; rr < 8; ++rr) {
          int rrow = m * 16 + (lane >> 4) * 8 + rr;
          int ccol = cb2 + n * 16 + (lane & 15);
          float val = acc[m][n][rr] + pT[rrow * PSTR + ccol];
          out[((b0 + rrow) * NH + head) * HD + ccol] = val;
        }
  }
}

// ---------------------------------------------------------------------------
extern "C" void kernel_launch(void* const* d_in, const int* in_sizes, int n_in,
                              void* d_out, int out_size, void* d_ws, size_t ws_size,
                              hipStream_t stream) {
  const float* x    = (const float*)d_in[0];
  const float* n1w  = (const float*)d_in[1];
  const float* n2w  = (const float*)d_in[2];
  const float* w1   = (const float*)d_in[3];
  const float* w3   = (const float*)d_in[4];
  const float* w2   = (const float*)d_in[5];
  const float* mask = (const float*)d_in[6];
  float* out = (float*)d_out;

  const size_t P_ELEMS  = (size_t)BATCHN * NH * HD;       // 33.5M f32
  const size_t WP_ELEMS = (size_t)NH * HD * FH;           // 2.1M bf16 each

  float*  p   = (float*)d_ws;
  __bf16* w1p = (__bf16*)((char*)d_ws + P_ELEMS * sizeof(float));
  __bf16* w3p = w1p + WP_ELEMS;
  __bf16* w2p = w3p + WP_ELEMS;

  // K0: pack weights to bf16 fragment layout (131072 lane-slots per tensor)
  pack_weights<<<dim3(512, 3), 256, 0, stream>>>(w1, w3, w2, w1p, w3p, w2p);
  // K1: rmsnorm1 + head mixing + mask + residual -> p
  mix_kernel<<<dim3(BATCHN), 256, 0, stream>>>(x, n1w, mask, p);
  // K2: TDM p-tile load + rmsnorm2 + SwiGLU FFN (bf16 WMMA) + residual -> out
  ffn_kernel<<<dim3(BATCHN / BM, NH), 256, 0, stream>>>(p, n2w, w1p, w3p, w2p, out);
}
